// SwinTransformerBlock_52046413692961
// MI455X (gfx1250) — matrix operand order repaired and checked by reading, not compiled, not run on hardware
//
#include <hip/hip_runtime.h>
#include <hip/hip_bf16.h>
#include <math.h>

// ---------------------------------------------------------------------------
// Types for CDNA5 WMMA (wave32): A/B = 16 halves per lane, C/D = 8 floats.
// ---------------------------------------------------------------------------
typedef _Float16 half_t;
typedef __attribute__((ext_vector_type(16))) _Float16 v16h;
typedef __attribute__((ext_vector_type(8)))  _Float16 h8;
typedef __attribute__((ext_vector_type(8)))  float    v8f;

#define B_      32
#define H_      56
#define W_      56
#define C_      512
#define WS_     7
#define SS_     3
#define NH_     16
#define NTOK    49
#define HD_     32
#define NWIN    64          // windows per image
#define MTOT    100352      // B*H*W tokens
#define HIDDEN_ 2048
#define SCALE_  0.17677669529663687f   // 32^-0.5

// ---------------------------------------------------------------------------
// Workspace layout (bytes, 256-aligned). Regions are reused across phases:
//   A1 region: LN1-out -> attn-out -> LN2-out (all MTOT x 512 f16)
//   R0 region: QKV (MTOT x 1536 f16) then MLP hidden (MTOT x 2048 f16)
//   BM: fused padded attention bias+mask [16][64][64][64] f32 (-1e9 pad cols)
// ---------------------------------------------------------------------------
static constexpr size_t OFF_WQKV  = 0;
static constexpr size_t OFF_WPROJ = OFF_WQKV  + (size_t)1536 * 512 * 2;
static constexpr size_t OFF_WFC1  = OFF_WPROJ + (size_t)512  * 512 * 2;
static constexpr size_t OFF_WFC2  = OFF_WFC1  + (size_t)2048 * 512 * 2;
static constexpr size_t OFF_BM    = OFF_WFC2  + (size_t)512  * 2048 * 2;
static constexpr size_t OFF_A1    = OFF_BM    + (size_t)16 * 64 * 64 * 64 * 4;
static constexpr size_t OFF_R0    = OFF_A1    + (size_t)MTOT * 512 * 2;
static constexpr size_t OFF_X2    = OFF_R0    + (size_t)MTOT * 2048 * 2;

// ---------------------------------------------------------------------------
// Helpers
// ---------------------------------------------------------------------------
__device__ __forceinline__ v16h mk16(h8 lo, h8 hiv) {
    v16h r;
#pragma unroll
    for (int e = 0; e < 8; ++e) { r[e] = lo[e]; r[e + 8] = hiv[e]; }
    return r;
}

// Window-order row (shifted image, window-partitioned) -> original-layout base
// index of that token's channel vector.  roll(h,-SS): shifted[y] = orig[(y+SS)%H]
__device__ __forceinline__ size_t win_row_to_gidx(int r) {
    int b  = r / 3136;
    int rw = r - b * 3136;
    int wi = rw / NTOK;
    int t  = rw - wi * NTOK;
    int wy = wi >> 3, wx = wi & 7;
    int y  = wy * WS_ + t / WS_;
    int x  = wx * WS_ + t % WS_;
    int ys = y + SS_; if (ys >= H_) ys -= H_;
    int xs = x + SS_; if (xs >= W_) xs -= W_;
    return (((size_t)b * H_ + ys) * W_ + xs) * (size_t)C_;
}

__device__ __forceinline__ int swin_region(int y, int x) {
    int ry = (y < H_ - WS_) ? 0 : ((y < H_ - SS_) ? 1 : 2);
    int rx = (x < W_ - WS_) ? 0 : ((x < W_ - SS_) ? 1 : 2);
    return ry * 3 + rx;
}

// ---------------------------------------------------------------------------
// Weight transpose + f16 convert:  src f32 [K][N] row-major -> dst f16 [N][K]
// (makes GEMM B fragments contiguous along K)
// ---------------------------------------------------------------------------
__global__ void w_transpose(const float* __restrict__ src, half_t* __restrict__ dst,
                            int K, int N) {
    int total = K * N;
    for (int idx = blockIdx.x * blockDim.x + threadIdx.x; idx < total;
         idx += gridDim.x * blockDim.x) {
        int k = idx / N, n = idx - k * N;
        dst[(size_t)n * K + k] = (half_t)src[idx];
    }
}

// ---------------------------------------------------------------------------
// Fused padded bias+mask: BM[h][wi][i][j] (i,j in [0,64)).
// Valid (i<49 && j<49): rel-pos bias + shift mask.  Else -1e9 (softmax pad).
// ---------------------------------------------------------------------------
__global__ void bm_kernel(const float* __restrict__ table, float* __restrict__ bm) {
    int idx = blockIdx.x * blockDim.x + threadIdx.x;
    if (idx >= 16 * 64 * 64 * 64) return;
    int j  = idx & 63;
    int i  = (idx >> 6) & 63;
    int wi = (idx >> 12) & 63;
    int h  = idx >> 18;
    float v = -1e9f;
    if (i < NTOK && j < NTOK) {
        int dy = (i / WS_ - j / WS_) + (WS_ - 1);
        int dx = (i % WS_ - j % WS_) + (WS_ - 1);
        float bias = table[(dy * (2 * WS_ - 1) + dx) * NH_ + h];
        int wy = wi >> 3, wx = wi & 7;
        int yi = wy * WS_ + i / WS_, xi = wx * WS_ + i % WS_;
        int yj = wy * WS_ + j / WS_, xj = wx * WS_ + j % WS_;
        float mask = (swin_region(yi, xi) != swin_region(yj, xj)) ? -100.0f : 0.0f;
        v = bias + mask;
    }
    bm[idx] = v;
}

// ---------------------------------------------------------------------------
// LayerNorm over C=512.  PERM=true: read via shifted-window permutation (LN1);
// PERM=false: natural token order (LN2).  Output f16 row-major A matrix.
// ---------------------------------------------------------------------------
template <bool PERM>
__global__ __launch_bounds__(128) void ln_kernel(const float* __restrict__ x,
                                                 const float* __restrict__ gamma,
                                                 const float* __restrict__ beta,
                                                 half_t* __restrict__ out) {
    __shared__ float r1[128], r2[128];
    int r = blockIdx.x;
    size_t gbase = PERM ? win_row_to_gidx(r) : (size_t)r * C_;
    int c0 = threadIdx.x * 4;
    float4 v = *(const float4*)(x + gbase + c0);
    r1[threadIdx.x] = v.x + v.y + v.z + v.w;
    r2[threadIdx.x] = v.x * v.x + v.y * v.y + v.z * v.z + v.w * v.w;
    __syncthreads();
    for (int st = 64; st > 0; st >>= 1) {
        if (threadIdx.x < st) {
            r1[threadIdx.x] += r1[threadIdx.x + st];
            r2[threadIdx.x] += r2[threadIdx.x + st];
        }
        __syncthreads();
    }
    float mu  = r1[0] * (1.0f / C_);
    float var = r2[0] * (1.0f / C_) - mu * mu;
    float rs  = rsqrtf(var + 1e-5f);
    float4 g4 = *(const float4*)(gamma + c0);
    float4 b4 = *(const float4*)(beta + c0);
    half_t* o = out + (size_t)r * C_ + c0;
    o[0] = (half_t)((v.x - mu) * rs * g4.x + b4.x);
    o[1] = (half_t)((v.y - mu) * rs * g4.y + b4.y);
    o[2] = (half_t)((v.z - mu) * rs * g4.z + b4.z);
    o[3] = (half_t)((v.w - mu) * rs * g4.w + b4.w);
}

// ---------------------------------------------------------------------------
// Tiled WMMA GEMM with double-buffered LDS:
//   C[M,N] = A[M,K](f16 row-major) * Bt[N,K](f16) + bias
// Block tile 128x128, K-step 32, 256 threads = 8 waves (2M x 4N); each wave
// owns 4x2 fragments -> 8 v_wmma_f32_16x16x32_f16 per K-step.  Next K-tile's
// global loads are issued before the WMMAs so latency hides under math.
// EPI: 0 bias->f16 | 1 bias+GELU->f16 | 2 bias+unshift-residual(x)->f32
//      3 bias+residual(X2)->f32
// ---------------------------------------------------------------------------
template <int EPI>
__global__ __launch_bounds__(256) void gemm_wmma(const half_t* __restrict__ A,
                                                 const half_t* __restrict__ Bt,
                                                 const float* __restrict__ bias,
                                                 const float* __restrict__ res,
                                                 void* __restrict__ outp,
                                                 int M, int N, int K) {
    __shared__ half_t As[2][128][40];   // pad stride 40 halves (80B, 16B-aligned)
    __shared__ half_t Bs[2][128][40];

    int tid  = threadIdx.x;
    int lane = tid & 31, wid = tid >> 5;
    int wm = wid >> 2, wn = wid & 3;
    int lr = lane & 15, hi = lane >> 4;       // A kb0 = hi*16, B ks = hi*16
    int rowBase = blockIdx.y * 128, colBase = blockIdx.x * 128;

    int ldr = tid >> 1, ldc = (tid & 1) * 16; // tile loader: 16 halves/thread
    const half_t* Ag = A  + (size_t)(rowBase + ldr) * K + ldc;
    const half_t* Bg = Bt + (size_t)(colBase + ldr) * K + ldc;

    v8f zero = {0.f, 0.f, 0.f, 0.f, 0.f, 0.f, 0.f, 0.f};
    v8f acc[4][2];
#pragma unroll
    for (int mf = 0; mf < 4; ++mf)
#pragma unroll
        for (int nf = 0; nf < 2; ++nf) acc[mf][nf] = zero;

    int T = K >> 5;
    {   // prologue: tile 0 -> buffer 0
        const h8* ap = (const h8*)Ag;
        *(h8*)&As[0][ldr][ldc]     = ap[0];
        *(h8*)&As[0][ldr][ldc + 8] = ap[1];
        const h8* bp = (const h8*)Bg;
        *(h8*)&Bs[0][ldr][ldc]     = bp[0];
        *(h8*)&Bs[0][ldr][ldc + 8] = bp[1];
    }
    __syncthreads();

    for (int it = 0; it < T; ++it) {
        int cur = it & 1, nxt = cur ^ 1;
        bool hasNext = (it + 1 < T);
        h8 ra0, ra1, rb0, rb1;
        if (hasNext) {          // issue next tile's global loads early
            const h8* ap = (const h8*)(Ag + (it + 1) * 32);
            ra0 = ap[0]; ra1 = ap[1];
            const h8* bp = (const h8*)(Bg + (it + 1) * 32);
            rb0 = bp[0]; rb1 = bp[1];
        }

        v16h a[4], b[2];
#pragma unroll
        for (int mf = 0; mf < 4; ++mf) {
            int row = wm * 64 + mf * 16 + lr;
            a[mf] = mk16(*(const h8*)&As[cur][row][hi * 16],
                         *(const h8*)&As[cur][row][hi * 16 + 8]);
        }
#pragma unroll
        for (int nf = 0; nf < 2; ++nf) {
            int col = wn * 32 + nf * 16 + lr;
            b[nf] = mk16(*(const h8*)&Bs[cur][col][hi * 16],
                         *(const h8*)&Bs[cur][col][hi * 16 + 8]);
        }
#pragma unroll
        for (int mf = 0; mf < 4; ++mf)
#pragma unroll
            for (int nf = 0; nf < 2; ++nf)
                acc[mf][nf] = __builtin_amdgcn_wmma_f32_16x16x32_f16(
                    false, a[mf], false, b[nf], (short)0, acc[mf][nf], false, false);

        if (hasNext) {          // park next tile while current was computing
            *(h8*)&As[nxt][ldr][ldc]     = ra0;
            *(h8*)&As[nxt][ldr][ldc + 8] = ra1;
            *(h8*)&Bs[nxt][ldr][ldc]     = rb0;
            *(h8*)&Bs[nxt][ldr][ldc + 8] = rb1;
        }
        __syncthreads();
    }

    // Epilogue: C layout = lane 0-15 row g, lane 16-31 row g+8, col = lane&15
#pragma unroll
    for (int mf = 0; mf < 4; ++mf) {
#pragma unroll
        for (int g = 0; g < 8; ++g) {
            int i = rowBase + wm * 64 + mf * 16 + g + hi * 8;
            size_t gb = 0;
            if (EPI == 2) gb = win_row_to_gidx(i);
#pragma unroll
            for (int nf = 0; nf < 2; ++nf) {
                int j = colBase + wn * 32 + nf * 16 + lr;
                float v = acc[mf][nf][g] + bias[j];
                if (EPI == 0) {
                    ((half_t*)outp)[(size_t)i * N + j] = (half_t)v;
                } else if (EPI == 1) {
                    float gl = 0.5f * v * (1.0f + erff(v * 0.70710678118654752f));
                    ((half_t*)outp)[(size_t)i * N + j] = (half_t)gl;
                } else if (EPI == 2) {
                    size_t gi = gb + j;
                    ((float*)outp)[gi] = res[gi] + v;
                } else {
                    size_t gi = (size_t)i * 512 + j;
                    ((float*)outp)[gi] = res[gi] + v;
                }
            }
        }
    }
}

// ---------------------------------------------------------------------------
// Windowed attention.  One 128-thread block (4 waves) per (window, head).
// S = (Q*SCALE) K^T + BM ; softmax ; O = P V, all on v_wmma f16.
// Tokens padded 49 -> 64; BM carries -1e9 in pad columns -> branchless.
// ---------------------------------------------------------------------------
__global__ __launch_bounds__(128) void attn_kernel(const half_t* __restrict__ qkv,
                                                   const float* __restrict__ bm,
                                                   half_t* __restrict__ outA) {
    __shared__ half_t Qs[64][40], Ks[64][40], Vs[64][40];
    __shared__ half_t Ps[4][16][72];

    int bx = blockIdx.x;
    int w = bx >> 4, h = bx & 15;
    int tid = threadIdx.x;
    int wid = tid >> 5, lane = tid & 31;
    int lr = lane & 15, hi = lane >> 4;
    size_t rowb = (size_t)w * NTOK;
    int qoff = h * HD_, koff = 512 + h * HD_, voff = 1024 + h * HD_;

    // stage Q (scaled), K, V as 16-byte chunks: 64 tokens x 4 chunks of 8 halves
    h8 zh = {0, 0, 0, 0, 0, 0, 0, 0};
    for (int c = tid; c < 256; c += 128) {
        int tok = c >> 2, d = (c & 3) * 8;
        h8 qv = zh, kv = zh, vv = zh;
        if (tok < NTOK) {
            const half_t* p = qkv + (rowb + tok) * 1536;
            qv = *(const h8*)(p + qoff + d);
#pragma unroll
            for (int e = 0; e < 8; ++e) qv[e] = (half_t)((float)qv[e] * SCALE_);
            kv = *(const h8*)(p + koff + d);
            vv = *(const h8*)(p + voff + d);
        }
        *(h8*)&Qs[tok][d] = qv;
        *(h8*)&Ks[tok][d] = kv;
        *(h8*)&Vs[tok][d] = vv;
    }
    __syncthreads();

    int mbase = wid * 16;
    int kb0 = hi * 16, ks = hi * 16;

    v16h aq;
    {
        int row = mbase + lr;
        aq = mk16(*(const h8*)&Qs[row][kb0], *(const h8*)&Qs[row][kb0 + 8]);
    }

    v8f zero = {0.f, 0.f, 0.f, 0.f, 0.f, 0.f, 0.f, 0.f};
    v8f sacc[4];
#pragma unroll
    for (int nf = 0; nf < 4; ++nf) {
        int col = nf * 16 + lr;
        v16h bk = mk16(*(const h8*)&Ks[col][ks], *(const h8*)&Ks[col][ks + 8]);
        sacc[nf] = __builtin_amdgcn_wmma_f32_16x16x32_f16(
            false, aq, false, bk, (short)0, zero, false, false);
    }

    // branchless bias+mask from fused padded plane BM[h][wi][64][64]
    int wi = w & (NWIN - 1);
    const float* bmp = bm + (((size_t)h * 64 + wi) << 12);
    float sv[4][8];
#pragma unroll
    for (int nf = 0; nf < 4; ++nf)
#pragma unroll
        for (int g = 0; g < 8; ++g) {
            int i = mbase + g + hi * 8;
            int j = nf * 16 + lr;
            sv[nf][g] = sacc[nf][g] + bmp[(i << 6) + j];
        }

    // row softmax: row elements live across 16 lanes (same hi half) x 4 frags
#pragma unroll
    for (int g = 0; g < 8; ++g) {
        float m = sv[0][g];
#pragma unroll
        for (int nf = 1; nf < 4; ++nf) m = fmaxf(m, sv[nf][g]);
        for (int off = 1; off < 16; off <<= 1) m = fmaxf(m, __shfl_xor(m, off, 16));
        float sum = 0.f;
#pragma unroll
        for (int nf = 0; nf < 4; ++nf) {
            float e = __expf(sv[nf][g] - m);
            sv[nf][g] = e;
            sum += e;
        }
        for (int off = 1; off < 16; off <<= 1) sum += __shfl_xor(sum, off, 16);
        float inv = 1.0f / sum;
#pragma unroll
        for (int nf = 0; nf < 4; ++nf) sv[nf][g] *= inv;
    }

    // stage P through LDS to re-shape C-layout -> A-fragment layout
#pragma unroll
    for (int nf = 0; nf < 4; ++nf)
#pragma unroll
        for (int g = 0; g < 8; ++g)
            Ps[wid][g + hi * 8][nf * 16 + lr] = (half_t)sv[nf][g];
    __syncthreads();

    v8f oacc[2];
    oacc[0] = zero; oacc[1] = zero;
#pragma unroll
    for (int kc = 0; kc < 2; ++kc) {
        v16h pa = mk16(*(const h8*)&Ps[wid][lr][kc * 32 + kb0],
                       *(const h8*)&Ps[wid][lr][kc * 32 + kb0 + 8]);
#pragma unroll
        for (int nf = 0; nf < 2; ++nf) {
            v16h vb;
#pragma unroll
            for (int e = 0; e < 16; ++e)
                vb[e] = Vs[kc * 32 + ks + e][nf * 16 + lr];
            oacc[nf] = __builtin_amdgcn_wmma_f32_16x16x32_f16(
                false, pa, false, vb, (short)0, oacc[nf], false, false);
        }
    }

#pragma unroll
    for (int nf = 0; nf < 2; ++nf)
#pragma unroll
        for (int g = 0; g < 8; ++g) {
            int i = mbase + g + hi * 8;
            if (i < NTOK)
                outA[(rowb + i) * (size_t)C_ + h * HD_ + nf * 16 + lr] =
                    (half_t)oacc[nf][g];
        }
}

// ---------------------------------------------------------------------------
// Launch
// ---------------------------------------------------------------------------
extern "C" void kernel_launch(void* const* d_in, const int* in_sizes, int n_in,
                              void* d_out, int out_size, void* d_ws, size_t ws_size,
                              hipStream_t stream) {
    (void)in_sizes; (void)n_in; (void)out_size; (void)ws_size;
    const float* x      = (const float*)d_in[0];
    const float* n1g    = (const float*)d_in[1];
    const float* n1b    = (const float*)d_in[2];
    const float* qkv_w  = (const float*)d_in[3];
    const float* qkv_b  = (const float*)d_in[4];
    const float* relt   = (const float*)d_in[5];
    const float* proj_w = (const float*)d_in[6];
    const float* proj_b = (const float*)d_in[7];
    const float* n2g    = (const float*)d_in[8];
    const float* n2b    = (const float*)d_in[9];
    const float* fc1_w  = (const float*)d_in[10];
    const float* fc1_b  = (const float*)d_in[11];
    const float* fc2_w  = (const float*)d_in[12];
    const float* fc2_b  = (const float*)d_in[13];

    char* ws = (char*)d_ws;
    half_t* Wqkv  = (half_t*)(ws + OFF_WQKV);
    half_t* Wproj = (half_t*)(ws + OFF_WPROJ);
    half_t* Wfc1  = (half_t*)(ws + OFF_WFC1);
    half_t* Wfc2  = (half_t*)(ws + OFF_WFC2);
    float*  BM    = (float*)(ws + OFF_BM);
    half_t* A1    = (half_t*)(ws + OFF_A1);   // LN1 out / attn out / LN2 out
    half_t* R0    = (half_t*)(ws + OFF_R0);   // QKV out, then MLP hidden
    float*  X2    = (float*)(ws + OFF_X2);    // residual-1 result (f32)
    float*  out   = (float*)d_out;

    // --- weight + fused bias/mask prep ---
    w_transpose<<<3072, 256, 0, stream>>>(qkv_w,  Wqkv, 512,  1536);
    w_transpose<<<1024, 256, 0, stream>>>(proj_w, Wproj, 512, 512);
    w_transpose<<<4096, 256, 0, stream>>>(fc1_w,  Wfc1, 512,  2048);
    w_transpose<<<4096, 256, 0, stream>>>(fc2_w,  Wfc2, 2048, 512);
    bm_kernel<<<(16 * 64 * 64 * 64) / 256, 256, 0, stream>>>(relt, BM);

    // --- LN1 + shift + window partition -> f16 A matrix ---
    ln_kernel<true><<<MTOT, 128, 0, stream>>>(x, n1g, n1b, A1);

    // --- QKV GEMM: (100352 x 512) @ (512 x 1536) ---
    gemm_wmma<0><<<dim3(1536 / 128, MTOT / 128), 256, 0, stream>>>(
        A1, Wqkv, qkv_b, nullptr, (void*)R0, MTOT, 1536, 512);

    // --- windowed attention (2048 windows x 16 heads) ---
    attn_kernel<<<2048 * 16, 128, 0, stream>>>(R0, BM, A1);

    // --- proj GEMM + window-reverse/unshift + residual(x) -> X2 f32 ---
    gemm_wmma<2><<<dim3(512 / 128, MTOT / 128), 256, 0, stream>>>(
        A1, Wproj, proj_b, x, (void*)X2, MTOT, 512, 512);

    // --- LN2 -> f16 A matrix ---
    ln_kernel<false><<<MTOT, 128, 0, stream>>>(X2, n2g, n2b, A1);

    // --- FC1 GEMM + GELU -> f16 hidden ---
    gemm_wmma<1><<<dim3(2048 / 128, MTOT / 128), 256, 0, stream>>>(
        A1, Wfc1, fc1_b, nullptr, (void*)R0, MTOT, 2048, 512);

    // --- FC2 GEMM + residual(X2) -> f32 output ---
    gemm_wmma<3><<<dim3(512 / 128, MTOT / 128), 256, 0, stream>>>(
        R0, Wfc2, fc2_b, X2, (void*)out, MTOT, 512, 2048);
}
